// CausalSelfAttention_23407571763355
// MI455X (gfx1250) — matrix operand order repaired and checked by reading, not compiled, not run on hardware
//
#include <hip/hip_runtime.h>
#include <hip/hip_bf16.h>
#include <math.h>
#include <stdint.h>

// ---------------------------------------------------------------------------
// Types for gfx1250 WMMA (wave32).
// ---------------------------------------------------------------------------
typedef __bf16 bf16_t;
typedef __attribute__((ext_vector_type(16))) __bf16 v16bf;
typedef __attribute__((ext_vector_type(4)))  __bf16 v4bf;
typedef __attribute__((ext_vector_type(8)))  float  v8f;

#define WMMA_BF16(a, b, c) \
    __builtin_amdgcn_wmma_f32_16x16x32_bf16(false, (a), false, (b), (short)0, (c), false, false)

__device__ __forceinline__ bf16_t f2bf(float f) { return (bf16_t)f; }

// gfx1250 async global->LDS copy (ASYNCcnt path).
__device__ __forceinline__ void async_ld_b128(void* lds_ptr, const void* gptr) {
    uint32_t lds_off = (uint32_t)(uintptr_t)lds_ptr;
    asm volatile("global_load_async_to_lds_b128 %0, %1, off"
                 :: "v"(lds_off), "v"(gptr) : "memory");
}
#if __has_builtin(__builtin_amdgcn_s_wait_asynccnt)
#define WAIT_ASYNC(n) __builtin_amdgcn_s_wait_asynccnt(n)
#else
#define WAIT_ASYNC(n) asm volatile("s_wait_asynccnt %0" :: "i"(n) : "memory")
#endif

// Reductions across a 16-lane half (C/D fragment rows live on 16-lane halves).
__device__ __forceinline__ float halfmax(float v) {
#pragma unroll
    for (int m = 1; m < 16; m <<= 1) v = fmaxf(v, __shfl_xor(v, m, 32));
    return v;
}
__device__ __forceinline__ float halfsum(float v) {
#pragma unroll
    for (int m = 1; m < 16; m <<= 1) v += __shfl_xor(v, m, 32);
    return v;
}

// ---------------------------------------------------------------------------
// Problem constants.
// ---------------------------------------------------------------------------
#define NB   4
#define SEQ  2048
#define NTOK (NB * SEQ)          // 8192
#define DIMM 1024
#define NH   16
#define NKVH 4
#define HD   64
#define KVD  (NKVH * HD)         // 256

// ---------------------------------------------------------------------------
// Pre-pass 1: vectorized f32 -> bf16 (x, v).
// ---------------------------------------------------------------------------
__global__ void cvt_f32_bf16_v4(const float4* __restrict__ in, v4bf* __restrict__ out, int n4)
{
    int i = blockIdx.x * blockDim.x + threadIdx.x;
    if (i < n4) {
        float4 v = in[i];
        v4bf r;
        r[0] = f2bf(v.x); r[1] = f2bf(v.y); r[2] = f2bf(v.z); r[3] = f2bf(v.w);
        out[i] = r;
    }
}

// ---------------------------------------------------------------------------
// Pre-pass 2: weight transpose + convert:  Wt[k][n] (bf16) = W[n][k] (f32).
// K == DIMM == 1024 for all weights here.
// ---------------------------------------------------------------------------
__global__ void transpose_cvt(const float* __restrict__ W, bf16_t* __restrict__ Wt, int N)
{
    int e = blockIdx.x * blockDim.x + threadIdx.x;
    if (e < N * DIMM) {
        int n = e >> 10, k = e & (DIMM - 1);
        Wt[(size_t)k * N + n] = f2bf(W[e]);
    }
}

// ---------------------------------------------------------------------------
// GEMM:  C[M,N] (f32) = A[M,K] (bf16 row-major) @ Wt[K,N] (bf16 k-major).
// Block = 256 thr = 8 waves; block tile 128x128; wave tile 32x64 (8 WMMAs per
// K-step). Double-buffered async global->LDS staging (ASYNCcnt): tile i+1
// streams in while tile i is consumed by the matrix pipe.
// ---------------------------------------------------------------------------
__global__ __launch_bounds__(256) void gemm_bf16(
    const bf16_t* __restrict__ A, const bf16_t* __restrict__ Wt,
    float* __restrict__ C, int M, int N, int K)
{
    __shared__ __attribute__((aligned(32))) bf16_t As[2][128][32];
    __shared__ __attribute__((aligned(32))) bf16_t Bs[2][32][128];

    const int t    = threadIdx.x;
    const int lane = t & 31;
    const int wave = t >> 5;
    const int wm   = wave >> 1;              // 4 row tiles of 32
    const int wn   = wave & 1;               // 2 col tiles of 64
    const int m0   = blockIdx.y * 128;
    const int n0   = blockIdx.x * 128;

    // Per-tile staging: 4 async b128 per thread (2 for A, 2 for B).
    const int arow = t >> 2, aseg = t & 3;
    auto stage = [&](int buf, int k0) {
        async_ld_b128(&As[buf][arow][aseg * 8],
                      A + (size_t)(m0 + arow) * K + k0 + aseg * 8);
        async_ld_b128(&As[buf][arow + 64][aseg * 8],
                      A + (size_t)(m0 + arow + 64) * K + k0 + aseg * 8);
#pragma unroll
        for (int i = 0; i < 2; ++i) {
            int e = i * 256 + t, kk = e >> 4, sg = e & 15;
            async_ld_b128(&Bs[buf][kk][sg * 8],
                          Wt + (size_t)(k0 + kk) * N + n0 + sg * 8);
        }
    };

    v8f acc[2][4] = {};

    stage(0, 0);
    int cur = 0;
    for (int k0 = 0; k0 < K; k0 += 32) {
        if (k0 + 32 < K) {          // prefetch next tile, wait only for current
            stage(cur ^ 1, k0 + 32);
            WAIT_ASYNC(4);
        } else {
            WAIT_ASYNC(0);
        }
        __syncthreads();            // copy -> compute

        v16bf a0 = *(const v16bf*)&As[cur][wm * 32      + (lane & 15)][(lane >> 4) * 16];
        v16bf a1 = *(const v16bf*)&As[cur][wm * 32 + 16 + (lane & 15)][(lane >> 4) * 16];
#pragma unroll
        for (int j = 0; j < 4; ++j) {
            v16bf b;
#pragma unroll
            for (int s = 0; s < 16; ++s)
                b[s] = Bs[cur][(lane >> 4) * 16 + s][wn * 64 + j * 16 + (lane & 15)];
            acc[0][j] = WMMA_BF16(a0, b, acc[0][j]);
            acc[1][j] = WMMA_BF16(a1, b, acc[1][j]);
        }
        cur ^= 1;
        __syncthreads();            // compute -> next overwrite
    }

    // C layout: VGPR g -> row = g + (lane/16)*8, col = lane%16.
#pragma unroll
    for (int i = 0; i < 2; ++i) {
#pragma unroll
        for (int j = 0; j < 4; ++j) {
            size_t base = (size_t)(m0 + wm * 32 + i * 16 + (lane >> 4) * 8) * N
                        + n0 + wn * 64 + j * 16 + (lane & 15);
#pragma unroll
            for (int g = 0; g < 8; ++g)
                C[base + (size_t)g * N] = acc[i][j][g];
        }
    }
}

// ---------------------------------------------------------------------------
// RMSNorm + RoPE for Q (row-major bf16 out, gain * 1/sqrt(HD) folded in).
// ---------------------------------------------------------------------------
__global__ __launch_bounds__(256) void rmsnorm_rope_q(
    const float* __restrict__ in, bf16_t* __restrict__ out,
    const float* __restrict__ gain)
{
    const int t    = threadIdx.x;
    const int lane = t & 31;
    const int w    = blockIdx.x * 8 + (t >> 5);
    const int tok  = w >> 4;                  // NH = 16
    const int h    = w & 15;
    const int s    = tok & (SEQ - 1);

    const size_t base = ((size_t)tok * NH + h) * HD;
    float t1 = in[base + lane];
    float t2 = in[base + 32 + lane];

    float ss = t1 * t1 + t2 * t2;
#pragma unroll
    for (int m = 1; m < 32; m <<= 1) ss += __shfl_xor(ss, m, 32);
    float r = rsqrtf(ss * (1.0f / 64.0f) + 1.1920929e-7f);
    t1 *= r; t2 *= r;

    float inv_freq = __expf(-0.28782313662f * (float)lane);  // ln(1e4)/32
    float sn, cs; __sincosf((float)s * inv_freq, &sn, &cs);
    float g  = gain[h] * 0.125f;              // fold 1/sqrt(64)
    float o1 = ( t1 * cs + t2 * sn) * g;
    float o2 = (-t1 * sn + t2 * cs) * g;

    out[base + lane]      = f2bf(o1);
    out[base + 32 + lane] = f2bf(o2);
}

// ---------------------------------------------------------------------------
// RMSNorm + RoPE for K, writing d-major (transposed) bf16: Kt[b][hkv][d][s].
// ---------------------------------------------------------------------------
__global__ __launch_bounds__(256) void rmsnorm_rope_k(
    const float* __restrict__ in, bf16_t* __restrict__ out_t)
{
    const int t    = threadIdx.x;
    const int lane = t & 31;
    const int w    = blockIdx.x * 8 + (t >> 5);
    const int tok  = w >> 2;                  // NKVH = 4
    const int h    = w & 3;
    const int b    = tok / SEQ;
    const int s    = tok - b * SEQ;

    const size_t base = ((size_t)tok * NKVH + h) * HD;
    float t1 = in[base + lane];
    float t2 = in[base + 32 + lane];

    float ss = t1 * t1 + t2 * t2;
#pragma unroll
    for (int m = 1; m < 32; m <<= 1) ss += __shfl_xor(ss, m, 32);
    float r = rsqrtf(ss * (1.0f / 64.0f) + 1.1920929e-7f);
    t1 *= r; t2 *= r;

    float inv_freq = __expf(-0.28782313662f * (float)lane);
    float sn, cs; __sincosf((float)s * inv_freq, &sn, &cs);
    float o1 =  t1 * cs + t2 * sn;
    float o2 = -t1 * sn + t2 * cs;

    const size_t tbase = ((size_t)(b * NKVH + h) * HD + lane) * SEQ + s;
    out_t[tbase]            = f2bf(o1);
    out_t[tbase + 32 * SEQ] = f2bf(o2);
}

// ---------------------------------------------------------------------------
// Causal flash attention. Block = 4 waves = one 64-row Q block of one (b,h).
// Double-buffered async K-tile staging; Q fragments are direct 32B global
// vector loads; V fragments are per-element global gathers. Output written as bf16 (feeds the projection GEMM).
// ---------------------------------------------------------------------------
__global__ __launch_bounds__(128) void flash_attn(
    const bf16_t* __restrict__ qb, const bf16_t* __restrict__ kt,
    const bf16_t* __restrict__ vb, bf16_t* __restrict__ yb)
{
    __shared__ __attribute__((aligned(32))) bf16_t Kt[2][HD][32];   // [buf][d][kv]
    __shared__ __attribute__((aligned(32))) bf16_t Pt[4][16][32];   // per-wave P (A-layout)

    const int t    = threadIdx.x;
    const int lane = t & 31;
    const int wave = t >> 5;
    const int qblk = blockIdx.x;
    const int b    = blockIdx.y >> 4;
    const int h    = blockIdx.y & 15;
    const int hkv  = h >> 2;
    const int qt0  = qblk * 64 + wave * 16;

    const bf16_t* kth = kt + (size_t)(b * NKVH + hkv) * HD * SEQ;

    // Per-chunk staging: 2 async b128 per thread (64 d-rows x 32 kv, 4KB).
    const int kd = t >> 1, ksg = t & 1;
    auto stage_k = [&](int buf, int kv0) {
        async_ld_b128(&Kt[buf][kd][ksg * 16],
                      kth + (size_t)kd * SEQ + kv0 + ksg * 16);
        async_ld_b128(&Kt[buf][kd][ksg * 16 + 8],
                      kth + (size_t)kd * SEQ + kv0 + ksg * 16 + 8);
    };

    // Q fragments, kept in registers for the whole kv loop.
    const size_t qrowbase =
        ((size_t)(b * SEQ + qt0 + (lane & 15)) * DIMM) + h * HD + (lane >> 4) * 16;
    const v16bf qa0 = *(const v16bf*)(qb + qrowbase);
    const v16bf qa1 = *(const v16bf*)(qb + qrowbase + 32);

    v8f o0 = {}, o1 = {}, o2 = {}, o3 = {};
    float mrow[8], lrow[8];
#pragma unroll
    for (int g = 0; g < 8; ++g) { mrow[g] = -INFINITY; lrow[g] = 0.0f; }

    const int kv_end = (qblk + 1) * 64;
    stage_k(0, 0);
    int cur = 0;
    for (int kv0 = 0; kv0 < kv_end; kv0 += 32) {
        if (kv0 + 32 < kv_end) {     // prefetch next K chunk
            stage_k(cur ^ 1, kv0 + 32);
            WAIT_ASYNC(2);
        } else {
            WAIT_ASYNC(0);
        }
        __syncthreads();             // copy -> compute

        // S = Q(16x64) @ K^T(64x32) as two 16x16 tiles, K split 2x32.
        v8f s0 = {}, s1 = {};
        {
            v16bf b00, b10, b01, b11;
#pragma unroll
            for (int s = 0; s < 16; ++s) {
                int d = (lane >> 4) * 16 + s;
                b00[s] = Kt[cur][d][lane & 15];
                b10[s] = Kt[cur][32 + d][lane & 15];
                b01[s] = Kt[cur][d][16 + (lane & 15)];
                b11[s] = Kt[cur][32 + d][16 + (lane & 15)];
            }
            s0 = WMMA_BF16(qa0, b00, s0);
            s0 = WMMA_BF16(qa1, b10, s0);
            s1 = WMMA_BF16(qa0, b01, s1);
            s1 = WMMA_BF16(qa1, b11, s1);
        }

        // Causal mask + online softmax (f32, half-wave row reductions).
        const int rbase = qt0 + (lane >> 4) * 8;
        const int c0    = kv0 + (lane & 15);
        float p0[8], p1[8];
#pragma unroll
        for (int g = 0; g < 8; ++g) {
            float a = s0[g], c = s1[g];
            if (c0      > rbase + g) a = -INFINITY;
            if (c0 + 16 > rbase + g) c = -INFINITY;
            float rmax  = halfmax(fmaxf(a, c));
            float mnew  = fmaxf(mrow[g], rmax);
            float alpha = __expf(mrow[g] - mnew);
            float ea = __expf(a - mnew);
            float ec = __expf(c - mnew);
            lrow[g] = lrow[g] * alpha + halfsum(ea + ec);
            mrow[g] = mnew;
            o0[g] *= alpha; o1[g] *= alpha; o2[g] *= alpha; o3[g] *= alpha;
            p0[g] = ea; p1[g] = ec;
        }

        // C-layout -> A-layout for P via per-wave LDS region.
#pragma unroll
        for (int g = 0; g < 8; ++g) {
            int pr = g + (lane >> 4) * 8;
            Pt[wave][pr][lane & 15]        = f2bf(p0[g]);
            Pt[wave][pr][16 + (lane & 15)] = f2bf(p1[g]);
        }
        __syncthreads();             // also fences this chunk's Kt reads
        v16bf pa = *(const v16bf*)&Pt[wave][lane & 15][(lane >> 4) * 16];

        // O += P(16x32) @ V(32x64): B fragment gathered as V[kv0 + 16*(lane/16) + s][d + lane%16].
        const size_t vbase = ((size_t)(b * SEQ + kv0) * KVD) + hkv * HD + (lane & 15);
        v16bf vf0, vf1, vf2, vf3;
#pragma unroll
        for (int s = 0; s < 16; ++s) {
            size_t vo = vbase + (size_t)((lane >> 4) * 16 + s) * KVD;
            vf0[s] = vb[vo]; vf1[s] = vb[vo + 16]; vf2[s] = vb[vo + 32]; vf3[s] = vb[vo + 48];
        }
        o0 = WMMA_BF16(pa, vf0, o0);
        o1 = WMMA_BF16(pa, vf1, o1);
        o2 = WMMA_BF16(pa, vf2, o2);
        o3 = WMMA_BF16(pa, vf3, o3);
        cur ^= 1;
    }

    // Normalize and store as bf16 (input to the projection GEMM).
#pragma unroll
    for (int g = 0; g < 8; ++g) {
        float inv = 1.0f / lrow[g];
        size_t row  = (size_t)(b * SEQ + qt0 + g + (lane >> 4) * 8);
        size_t base = row * DIMM + h * HD + (lane & 15);
        yb[base]      = f2bf(o0[g] * inv);
        yb[base + 16] = f2bf(o1[g] * inv);
        yb[base + 32] = f2bf(o2[g] * inv);
        yb[base + 48] = f2bf(o3[g] * inv);
    }
}

// ---------------------------------------------------------------------------
// Host-side launcher.
// ---------------------------------------------------------------------------
extern "C" void kernel_launch(void* const* d_in, const int* in_sizes, int n_in,
                              void* d_out, int out_size, void* d_ws, size_t ws_size,
                              hipStream_t stream)
{
    (void)in_sizes; (void)n_in; (void)out_size; (void)ws_size;
    const float* x     = (const float*)d_in[0];
    const float* Wq    = (const float*)d_in[1];
    const float* Wk    = (const float*)d_in[2];
    const float* Wv    = (const float*)d_in[3];
    const float* Wproj = (const float*)d_in[4];
    const float* qgain = (const float*)d_in[5];
    float* out = (float*)d_out;

    // Workspace carve-up (~115 MB).
    char* ws = (char*)d_ws;
    bf16_t* x_b  = (bf16_t*)ws;  ws += (size_t)NTOK * DIMM * 2;
    bf16_t* Wq_t = (bf16_t*)ws;  ws += (size_t)DIMM * DIMM * 2;
    bf16_t* Wk_t = (bf16_t*)ws;  ws += (size_t)KVD  * DIMM * 2;
    bf16_t* Wv_t = (bf16_t*)ws;  ws += (size_t)KVD  * DIMM * 2;
    bf16_t* Wp_t = (bf16_t*)ws;  ws += (size_t)DIMM * DIMM * 2;
    float*  q_f  = (float*)ws;   ws += (size_t)NTOK * DIMM * 4;
    float*  k_f  = (float*)ws;   ws += (size_t)NTOK * KVD  * 4;
    float*  v_f  = (float*)ws;   ws += (size_t)NTOK * KVD  * 4;
    bf16_t* q_b  = (bf16_t*)ws;  ws += (size_t)NTOK * DIMM * 2;
    bf16_t* kt_b = (bf16_t*)ws;  ws += (size_t)NTOK * KVD  * 2;
    bf16_t* v_b  = (bf16_t*)ws;  ws += (size_t)NTOK * KVD  * 2;
    bf16_t* y_b  = (bf16_t*)ws;  ws += (size_t)NTOK * DIMM * 2;

    // 0) One-time conversions / weight transposes.
    cvt_f32_bf16_v4<<<(NTOK * DIMM / 4 + 255) / 256, 256, 0, stream>>>(
        (const float4*)x, (v4bf*)x_b, NTOK * DIMM / 4);
    transpose_cvt<<<(DIMM * DIMM + 255) / 256, 256, 0, stream>>>(Wq,    Wq_t, DIMM);
    transpose_cvt<<<(KVD  * DIMM + 255) / 256, 256, 0, stream>>>(Wk,    Wk_t, KVD);
    transpose_cvt<<<(KVD  * DIMM + 255) / 256, 256, 0, stream>>>(Wv,    Wv_t, KVD);
    transpose_cvt<<<(DIMM * DIMM + 255) / 256, 256, 0, stream>>>(Wproj, Wp_t, DIMM);

    // 1) QKV projections (double-buffered async WMMA GEMMs).
    gemm_bf16<<<dim3(DIMM / 128, NTOK / 128), 256, 0, stream>>>(x_b, Wq_t, q_f, NTOK, DIMM, DIMM);
    gemm_bf16<<<dim3(KVD  / 128, NTOK / 128), 256, 0, stream>>>(x_b, Wk_t, k_f, NTOK, KVD,  DIMM);
    gemm_bf16<<<dim3(KVD  / 128, NTOK / 128), 256, 0, stream>>>(x_b, Wv_t, v_f, NTOK, KVD,  DIMM);

    // 2) RMSNorm + RoPE; q row-major (gain/scale folded), k d-major transposed.
    rmsnorm_rope_q<<<(NTOK * NH)   / 8, 256, 0, stream>>>(q_f, q_b, qgain);
    rmsnorm_rope_k<<<(NTOK * NKVH) / 8, 256, 0, stream>>>(k_f, kt_b);

    // 3) V -> bf16 (vectorized).
    cvt_f32_bf16_v4<<<(NTOK * KVD / 4 + 255) / 256, 256, 0, stream>>>(
        (const float4*)v_f, (v4bf*)v_b, NTOK * KVD / 4);

    // 4) Causal flash attention (WMMA + double-buffered async K staging).
    flash_attn<<<dim3(SEQ / 64, NB * NH), 128, 0, stream>>>(q_b, kt_b, v_b, y_b);

    // 5) Output projection straight into d_out (f32).
    gemm_bf16<<<dim3(DIMM / 128, NTOK / 128), 256, 0, stream>>>(y_b, Wp_t, out, NTOK, DIMM, DIMM);
}
